// Nystromformer_9947144258103
// MI455X (gfx1250) — matrix operand order repaired and checked by reading, not compiled, not run on hardware
//
#include <hip/hip_runtime.h>
#include <hip/hip_bf16.h>
#include <math.h>

// ---------------- problem constants ----------------
#define BATCH 4
#define HEADS 12
#define BH    48          // BATCH*HEADS
#define SEQ   4096
#define DIM   64
#define NLM   64          // landmarks
#define SEG   64          // SEQ / NLM
#define SCALE 0.125f      // 1/sqrt(64)
#define SPLITS 8
#define KEYS_PER_SPLIT (SEQ / SPLITS)   // 512
#define CHUNK 128
#define NCHUNK (KEYS_PER_SPLIT / CHUNK) // 4

typedef __attribute__((ext_vector_type(16))) _Float16 v16h;
typedef __attribute__((ext_vector_type(8)))  _Float16 v8h;
typedef __attribute__((ext_vector_type(8)))  float    v8f;
typedef __attribute__((ext_vector_type(2)))  float    v2f;
typedef __attribute__((ext_vector_type(4)))  unsigned tdm4;
typedef __attribute__((ext_vector_type(8)))  unsigned tdm8;

#if defined(__has_builtin)
#if __has_builtin(__builtin_amdgcn_wmma_f32_16x16x4_f32)
#define PINV_WMMA 1
#endif
#endif

// ---------------- async global->LDS copy (CDNA5 ASYNCcnt path) ----------------
// Copies nrows rows of 64 halves from gsrc (row stride 64) into LDS (row stride
// 72 halves). Each lane issues one GLOBAL_LOAD_ASYNC_TO_LDS_B128 with a
// per-lane LDS destination (ISA 10/15.18).
// Generic LDS pointer -> 32-bit LDS offset by truncation.
__device__ inline void async_copy_64_to_72(const _Float16* gsrc, _Float16* ldst,
                                           int t, int nrows) {
  int total = nrows * 8;                    // number of 16-byte chunks
  for (int i = t; i < total; i += 256) {
    int r = i >> 3, c = (i & 7) * 8;
    unsigned long long g = (unsigned long long)(uintptr_t)(gsrc + r * 64 + c);
    unsigned l = (unsigned)(uintptr_t)(ldst + r * 72 + c);
    asm volatile("global_load_async_to_lds_b128 %0, %1, off"
                 :: "v"(l), "v"(g) : "memory");
  }
}
__device__ inline void async_wait_all() {
  asm volatile("s_wait_asynccnt 0x0" ::: "memory");
}

// ---------------- Tensor Data Mover: 2D fp32 tile -> packed LDS ----------------
// Builds a D# (ISA 8.3/8.4) for a [rows x 64] fp32 tile with contiguous rows
// and issues TENSOR_LOAD_TO_LDS. Caller must be wave-uniform and gate to a
// single wave (TDM ignores EXEC; one DMA is issued per executing wave), then
// tdm_wait() + __syncthreads() before consuming the LDS tile.
__device__ inline void tdm_load_2d_f32(const float* gsrc, float* lds_dst,
                                       unsigned rows, unsigned row_elems) {
  unsigned long long ga = (unsigned long long)(uintptr_t)gsrc;
  unsigned lds = (unsigned)(uintptr_t)lds_dst;
  tdm4 g0;
  g0[0] = 1u;                                           // count=1, user mode
  g0[1] = lds;                                          // lds_addr (bytes)
  g0[2] = (unsigned)(ga & 0xFFFFFFFFu);                 // global_addr[31:0]
  g0[3] = (unsigned)((ga >> 32) & 0x01FFFFFFu)          // global_addr[56:32]
        | (2u << 30);                                   // type = 2 ("image")
  tdm8 g1;
  g1[0] = (2u << 16);                                   // data_size = 4B; mask=0
  g1[1] = (row_elems & 0xFFFFu) << 16;                  // tensor_dim0[15:0]
  g1[2] = ((row_elems >> 16) & 0xFFFFu)                 // tensor_dim0[31:16]
        | ((rows & 0xFFFFu) << 16);                     // tensor_dim1[15:0]
  g1[3] = ((rows >> 16) & 0xFFFFu)                      // tensor_dim1[31:16]
        | ((row_elems & 0xFFFFu) << 16);                // tile_dim0
  g1[4] = rows & 0xFFFFu;                               // tile_dim1 (tile_dim2=0)
  g1[5] = row_elems;                                    // tensor_dim0_stride[31:0]
  g1[6] = (row_elems & 0xFFFFu) << 16;                  // d0s[47:32]=0 | d1s[15:0]
  g1[7] = (row_elems >> 16) & 0xFFFFu;                  // tensor_dim1_stride[47:16]
  tdm4 gz = {0u, 0u, 0u, 0u};                           // 2D tensor: groups 2/3 unused
  asm volatile("tensor_load_to_lds %0, %1, %2, %3"
               :: "s"(g0), "s"(g1), "s"(gz), "s"(gz) : "memory");
}
__device__ inline void tdm_wait() {
  asm volatile("s_wait_tensorcnt 0x0" ::: "memory");
}

// ---------------- WMMA fragment helpers (wave32, 16x16 tiles) ----------------
__device__ inline v8h ld8h(const _Float16* p) { return *(const v8h*)p; }

// A (16x32 f16) stored row-major [m][k], ld must be a multiple of 8 halves.
__device__ inline v16h frag_a(const _Float16* p, int ld, int r0, int c0, int lane) {
  int m = r0 + (lane & 15);
  int g = (lane >> 4) & 1;
  const _Float16* base = p + m * ld + c0 + 8 * g;
  v8h lo = ld8h(base);
  v8h hi = ld8h(base + 16);
  return __builtin_shufflevector(lo, hi, 0, 1, 2, 3, 4, 5, 6, 7,
                                 8, 9, 10, 11, 12, 13, 14, 15);
}
// B (32x16 f16) where the operand is stored transposed, i.e. storage[n][k].
__device__ inline v16h frag_btr(const _Float16* p, int ld, int k0, int n0, int lane) {
  int n = n0 + (lane & 15);
  int g = (lane >> 4) & 1;
  const _Float16* base = p + n * ld + k0 + 8 * g;
  v8h lo = ld8h(base);
  v8h hi = ld8h(base + 16);
  return __builtin_shufflevector(lo, hi, 0, 1, 2, 3, 4, 5, 6, 7,
                                 8, 9, 10, 11, 12, 13, 14, 15);
}
__device__ inline v8f wmma16(v16h a, v16h b, v8f c) {
  return __builtin_amdgcn_wmma_f32_16x16x32_f16(false, a, false, b, (short)0, c,
                                                false, false);
}

// ---------------- K0: init reduction slots ----------------
__global__ void k_init(unsigned* coef) {
  if (threadIdx.x == 0) { coef[0] = 0u; coef[1] = 0u; }
}

// ---------------- K1: landmark (segment) means ----------------
__global__ void k_landmarks(const float* __restrict__ Q, const float* __restrict__ K,
                            float* qlm, float* klm, _Float16* qlmh, _Float16* klmh) {
  int bh = blockIdx.x;
  int t  = threadIdx.x;
  size_t off = (size_t)bh * SEQ * DIM;
#pragma unroll
  for (int which = 0; which < 2; ++which) {
    const float* src = which ? (K + off) : (Q + off);
    float*      dst  = which ? (klm + bh * NLM * DIM) : (qlm + bh * NLM * DIM);
    _Float16*   dsth = which ? (klmh + bh * NLM * DIM) : (qlmh + bh * NLM * DIM);
    for (int base = 0; base < NLM; base += 4) {   // 4 landmarks x 64 dims = 256 thr
      int lm = base + (t >> 6);
      int d  = t & 63;
      const float* p = src + (size_t)(lm * SEG) * DIM + d;
      float s = 0.f;
#pragma unroll 8
      for (int r = 0; r < SEG; ++r) s += p[r * DIM];
      s *= (1.0f / (float)SEG);
      dst[lm * DIM + d]  = s;
      dsth[lm * DIM + d] = (_Float16)s;
    }
  }
}

// ---------------- K2: softmax(q_lm k_lm^T) + global init_coef reduction ---------
__global__ void k_lm_attn(const float* qlm, const float* klm, float* A0,
                          unsigned* coef) {
  __shared__ float sq[NLM][DIM];
  __shared__ float sk[NLM][DIM];
  __shared__ float sm[NLM][NLM];
  int bh = blockIdx.x, t = threadIdx.x;
  for (int i = t; i < NLM * DIM; i += 256) {
    sq[i >> 6][i & 63] = qlm[bh * 4096 + i];
    sk[i >> 6][i & 63] = klm[bh * 4096 + i];
  }
  __syncthreads();
  for (int i = t; i < NLM * NLM; i += 256) {
    int r = i >> 6, c = i & 63;
    float s = 0.f;
#pragma unroll 8
    for (int d = 0; d < DIM; ++d) s += sq[r][d] * sk[c][d];
    sm[r][c] = s * SCALE;
  }
  __syncthreads();
  if (t < NLM) {                      // row softmax
    float mx = -1e30f;
    for (int c = 0; c < NLM; ++c) mx = fmaxf(mx, sm[t][c]);
    float sum = 0.f;
    for (int c = 0; c < NLM; ++c) { float e = __expf(sm[t][c] - mx); sm[t][c] = e; sum += e; }
    float inv = 1.0f / sum;
    for (int c = 0; c < NLM; ++c) sm[t][c] *= inv;
  }
  __syncthreads();
  for (int i = t; i < NLM * NLM; i += 256) A0[bh * 4096 + i] = sm[i >> 6][i & 63];
  if (t < NLM) {                      // abs-col-sum (axis=-2) and abs-row-sum (axis=-1)
    float cs = 0.f, rs = 0.f;
    for (int r = 0; r < NLM; ++r) cs += sm[r][t];
    for (int c = 0; c < NLM; ++c) rs += sm[t][c];
    atomicMax(coef + 0, __float_as_uint(cs));  // values >= 0: uint order == float order
    atomicMax(coef + 1, __float_as_uint(rs));
  }
}

// ---------------- 64x64 fp32 matmul in LDS (WMMA f32 16x16x4) ----------------
__device__ void mm64(const float* A, const float* B, float* C) {
  __syncthreads();                     // inputs ready
  int lane = threadIdx.x & 31, wv = threadIdx.x >> 5;
#ifdef PINV_WMMA
  int m = lane & 15, g = (lane >> 4) & 1;
  for (int ti = wv; ti < 16; ti += 8) {
    int mt = ti >> 2, nt = ti & 3;
    v8f acc = {};
#pragma unroll
    for (int kk = 0; kk < 16; ++kk) {
      v2f a, b;
      a[0] = A[(mt * 16 + m) * 64 + kk * 4 + 2 * g + 0];
      a[1] = A[(mt * 16 + m) * 64 + kk * 4 + 2 * g + 1];
      b[0] = B[(kk * 4 + 2 * g + 0) * 64 + nt * 16 + m];
      b[1] = B[(kk * 4 + 2 * g + 1) * 64 + nt * 16 + m];
      acc = __builtin_amdgcn_wmma_f32_16x16x4_f32(false, a, false, b, (short)0,
                                                  acc, false, false);
    }
#pragma unroll
    for (int v = 0; v < 8; ++v)
      C[(mt * 16 + v + 8 * g) * 64 + nt * 16 + m] = acc[v];
  }
#else
  (void)lane; (void)wv;
  for (int i = threadIdx.x; i < 4096; i += 256) {
    int r = i >> 6, c = i & 63;
    float s = 0.f;
    for (int d = 0; d < 64; ++d) s += A[r * 64 + d] * B[d * 64 + c];
    C[i] = s;
  }
#endif
  __syncthreads();                     // outputs visible
}

__device__ inline void sub_from_diag(float* Mb, float diag) {
  for (int i = threadIdx.x; i < 4096; i += 256) {
    int r = i >> 6, c = i & 63;
    Mb[i] = ((r == c) ? diag : 0.0f) - Mb[i];
  }
}

// ---------------- K3: Newton-Schulz pseudo-inverse ----------------
__global__ void k_pinv(const float* A0, const unsigned* coef, float* PINV) {
  extern __shared__ float sm3[];       // 4 x 64x64 fp32 = 64 KB dynamic LDS
  float* mat  = sm3;
  float* outb = sm3 + 4096;
  float* t1   = sm3 + 8192;
  float* t2   = sm3 + 12288;
  int bh = blockIdx.x, t = threadIdx.x;
  float coefv = __uint_as_float(coef[0]) * __uint_as_float(coef[1]);
  float inv   = 1.0f / coefv;
  for (int i = t; i < 4096; i += 256) mat[i] = A0[bh * 4096 + i];
  __syncthreads();
  for (int i = t; i < 4096; i += 256) {  // out = mat^T / coef
    int r = i >> 6, c = i & 63;
    outb[i] = mat[c * 64 + r] * inv;
  }
  for (int it = 0; it < 6; ++it) {
    mm64(mat, outb, t1);  sub_from_diag(t1, 7.0f);    // term1
    mm64(outb, t1, t2);
    mm64(mat, t2, t1);    sub_from_diag(t1, 15.0f);   // term2
    mm64(outb, t1, t2);
    mm64(mat, t2, t1);    sub_from_diag(t1, 13.0f);   // term3
    mm64(outb, t1, t2);
    __syncthreads();
    for (int i = t; i < 4096; i += 256) outb[i] = 0.25f * t2[i];
    __syncthreads();
  }
  for (int i = t; i < 4096; i += 256) PINV[bh * 4096 + i] = outb[i];
}

// ---------------- K4: B_tilde @ V, split flash-style online softmax ----------
// LDS (bytes): sq 9216 | sk 18432 | svT 17408 | Sb 33792 | Pb 17408 | ml 768
//              | stageK 32768 | stageV 32768   = 162560
#define K4_LDS (9216 + 18432 + 17408 + 33792 + 17408 + 768 + 65536)
__global__ void k_bv_partial(const float* __restrict__ Kg, const float* __restrict__ Vg,
                             const _Float16* qlmh, float* BVp, float* MLp) {
  extern __shared__ char smem[];
  _Float16* sq     = (_Float16*)smem;               // 64 x 72 (q_lm)
  _Float16* sk     = sq + 64 * 72;                  // 128 x 72 ([key][d])
  _Float16* svT    = sk + 128 * 72;                 // 64 x 136 ([d][key])
  float*    Sb     = (float*)(svT + 64 * 136);      // 64 x 132
  _Float16* Pb     = (_Float16*)(Sb + 64 * 132);    // 64 x 136
  float*    mrun   = (float*)(Pb + 64 * 136);       // 64
  float*    lrun   = mrun + 64;                     // 64
  float*    fact   = lrun + 64;                     // 64
  float*    stageK = (float*)(smem + 97024);        // 128 x 64 fp32 (TDM dest)
  float*    stageV = stageK + CHUNK * DIM;          // 128 x 64 fp32 (TDM dest)
  int sp = blockIdx.x, bh = blockIdx.y, t = threadIdx.x;
  int lane = t & 31, wv = t >> 5;
  const float* Kb = Kg + (size_t)bh * SEQ * DIM;
  const float* Vb = Vg + (size_t)bh * SEQ * DIM;
  async_copy_64_to_72(qlmh + bh * 4096, sq, t, 64);   // async DMA, no conversion
  if (t < 64) { mrun[t] = -1e30f; lrun[t] = 0.f; }
  int ti0 = wv, ti1 = wv + 8;                     // the 2 output tiles of this wave
  int mt0 = ti0 >> 2, nt0 = ti0 & 3, mt1 = ti1 >> 2, nt1 = ti1 & 3;
  v8f acc0 = {}; v8f acc1 = {};
  async_wait_all();
  __syncthreads();
  for (int ch = 0; ch < NCHUNK; ++ch) {
    int key0 = sp * KEYS_PER_SPLIT + ch * CHUNK;
    // K and V chunk tiles via Tensor Data Mover (wave 0 issues; TENSORcnt)
    if (wv == 0) {
      tdm_load_2d_f32(Kb + (size_t)key0 * DIM, stageK, CHUNK, DIM);
      tdm_load_2d_f32(Vb + (size_t)key0 * DIM, stageV, CHUNK, DIM);
      tdm_wait();
    }
    __syncthreads();
    for (int i = t; i < CHUNK * DIM; i += 256) {   // fp32 -> f16 operand layouts
      int r = i >> 6, c = i & 63;
      sk[r * 72 + c]   = (_Float16)stageK[i];
      svT[c * 136 + r] = (_Float16)stageV[i];
    }
    __syncthreads();
    // GEMM1: S[64x128] = qlm @ Kc^T ; wave wv owns N-tile wv
    {
      int nt = wv, g = (lane >> 4) & 1, n = lane & 15;
      for (int mt = 0; mt < 4; ++mt) {
        v8f cc = {};
#pragma unroll
        for (int ks = 0; ks < 2; ++ks) {
          v16h a = frag_a(sq, 72, mt * 16, ks * 32, lane);
          v16h b = frag_btr(sk, 72, ks * 32, nt * 16, lane);
          cc = wmma16(a, b, cc);
        }
#pragma unroll
        for (int v = 0; v < 8; ++v)
          Sb[(mt * 16 + v + 8 * g) * 132 + nt * 16 + n] = cc[v] * SCALE;
      }
    }
    __syncthreads();
    if (t < 64) {                      // online softmax on row t
      float rm = -1e30f;
      for (int c = 0; c < CHUNK; ++c) rm = fmaxf(rm, Sb[t * 132 + c]);
      float nm = fmaxf(mrun[t], rm);
      float f  = __expf(mrun[t] - nm);
      float ls = 0.f;
      for (int c = 0; c < CHUNK; ++c) {
        float e = __expf(Sb[t * 132 + c] - nm);
        Pb[t * 136 + c] = (_Float16)e;
        ls += e;
      }
      lrun[t] = lrun[t] * f + ls;
      mrun[t] = nm;
      fact[t] = f;
    }
    __syncthreads();
    // rescale register accumulators, then O += P @ Vc (K = 128 -> 4 steps)
    {
      int g = (lane >> 4) & 1;
#pragma unroll
      for (int v = 0; v < 8; ++v) {
        acc0[v] *= fact[mt0 * 16 + v + 8 * g];
        acc1[v] *= fact[mt1 * 16 + v + 8 * g];
      }
#pragma unroll
      for (int ks = 0; ks < 4; ++ks) {
        v16h a0 = frag_a(Pb, 136, mt0 * 16, ks * 32, lane);
        v16h b0 = frag_btr(svT, 136, ks * 32, nt0 * 16, lane);
        acc0 = wmma16(a0, b0, acc0);
        v16h a1 = frag_a(Pb, 136, mt1 * 16, ks * 32, lane);
        v16h b1 = frag_btr(svT, 136, ks * 32, nt1 * 16, lane);
        acc1 = wmma16(a1, b1, acc1);
      }
    }
    __syncthreads();
  }
  // write unnormalized partial O, running (m, l)
  {
    int g = (lane >> 4) & 1, n = lane & 15;
    float* dst = BVp + (size_t)(sp * BH + bh) * 4096;
#pragma unroll
    for (int v = 0; v < 8; ++v) {
      dst[(mt0 * 16 + v + 8 * g) * 64 + nt0 * 16 + n] = acc0[v];
      dst[(mt1 * 16 + v + 8 * g) * 64 + nt1 * 16 + n] = acc1[v];
    }
  }
  if (t < 64) {
    MLp[((size_t)(sp * BH + bh) * 64 + t) * 2 + 0] = mrun[t];
    MLp[((size_t)(sp * BH + bh) * 64 + t) * 2 + 1] = lrun[t];
  }
}

// ---------------- K4b: combine splits, then M = pinv @ BV (store M^T) --------
__global__ void k_bv_combine(const float* BVp, const float* MLp, const float* PINV,
                             float* MM, _Float16* MMhT) {
  __shared__ float wsp[SPLITS][64];
  __shared__ float Ls[64];
  __shared__ float bv[64][65];
  __shared__ float pv[64][65];
  int bh = blockIdx.x, t = threadIdx.x;
  if (t < 64) {
    float gm = -1e30f;
    for (int sp = 0; sp < SPLITS; ++sp)
      gm = fmaxf(gm, MLp[((size_t)(sp * BH + bh) * 64 + t) * 2 + 0]);
    float L = 0.f;
    for (int sp = 0; sp < SPLITS; ++sp) {
      float m = MLp[((size_t)(sp * BH + bh) * 64 + t) * 2 + 0];
      float l = MLp[((size_t)(sp * BH + bh) * 64 + t) * 2 + 1];
      float w = __expf(m - gm);
      wsp[sp][t] = w;
      L += l * w;
    }
    Ls[t] = L;
  }
  for (int i = t; i < 4096; i += 256) pv[i >> 6][i & 63] = PINV[bh * 4096 + i];
  __syncthreads();
  for (int i = t; i < 4096; i += 256) {
    int r = i >> 6, c = i & 63;
    float s = 0.f;
    for (int sp = 0; sp < SPLITS; ++sp)
      s += BVp[(size_t)(sp * BH + bh) * 4096 + i] * wsp[sp][r];
    bv[r][c] = s / Ls[r];
  }
  __syncthreads();
  for (int i = t; i < 4096; i += 256) {
    int r = i >> 6, c = i & 63;      // r = landmark (k), c = output dim (n)
    float s = 0.f;
#pragma unroll 8
    for (int d = 0; d < 64; ++d) s += pv[r][d] * bv[d][c];
    MM[bh * 4096 + i]            = s;
    MMhT[bh * 4096 + c * 64 + r] = (_Float16)s;   // transposed [n][k]
  }
}

// ---------------- K5: out = softmax(Q k_lm^T) @ M, fused ----------------
// LDS (bytes): sQ 18432 | sK 9216 | sMt 9216 | Sb 34816 | Pb 18432 | stage 32768
#define K5_LDS (18432 + 9216 + 9216 + 34816 + 18432 + 32768)
__global__ void k_final(const float* __restrict__ Qg, const _Float16* klmh,
                        const _Float16* MMhT, float* __restrict__ Out) {
  extern __shared__ char smem[];
  _Float16* sQ    = (_Float16*)smem;              // 128 x 72 (f16 of Q tile)
  _Float16* sK    = sQ + 128 * 72;                // 64 x 72  ([lm][d])
  _Float16* sMt   = sK + 64 * 72;                 // 64 x 72  ([n][k] = M^T)
  float*    Sb    = (float*)(sMt + 64 * 72);      // 128 x 68 (also output staging)
  _Float16* Pb    = (_Float16*)(Sb + 128 * 68);   // 128 x 72
  float*    stage = (float*)(smem + 90112);       // 128 x 64 fp32 (TDM dest)
  int tile = blockIdx.x, bh = blockIdx.y, t = threadIdx.x;
  int lane = t & 31, wv = t >> 5;
  const float* Qb = Qg + ((size_t)bh * SEQ + (size_t)tile * 128) * DIM;
  async_copy_64_to_72(klmh + bh * 4096, sK, t, 64);   // ASYNCcnt DMA paths
  async_copy_64_to_72(MMhT + bh * 4096, sMt, t, 64);
  if (wv == 0) {                                      // TENSORcnt DMA path
    tdm_load_2d_f32(Qb, stage, 128, DIM);
    tdm_wait();
  }
  async_wait_all();
  __syncthreads();
  for (int i = t; i < 128 * 64; i += 256) sQ[(i >> 6) * 72 + (i & 63)] = (_Float16)stage[i];
  __syncthreads();
  // GEMM1: S = Q_tile @ k_lm^T ; wave wv owns M-tile wv (rows 16wv..16wv+15)
  {
    int g = (lane >> 4) & 1, n = lane & 15;
    for (int nt = 0; nt < 4; ++nt) {
      v8f acc = {};
#pragma unroll
      for (int ks = 0; ks < 2; ++ks) {
        v16h a = frag_a(sQ, 72, wv * 16, ks * 32, lane);
        v16h b = frag_btr(sK, 72, ks * 32, nt * 16, lane);
        acc = wmma16(a, b, acc);
      }
#pragma unroll
      for (int v = 0; v < 8; ++v)
        Sb[(wv * 16 + v + 8 * g) * 68 + nt * 16 + n] = acc[v] * SCALE;
    }
  }
  __syncthreads();
  if (t < 128) {                      // full row softmax over 64 landmarks
    float mx = -1e30f;
    for (int c = 0; c < 64; ++c) mx = fmaxf(mx, Sb[t * 68 + c]);
    float s = 0.f;
    for (int c = 0; c < 64; ++c) { float e = __expf(Sb[t * 68 + c] - mx); Sb[t * 68 + c] = e; s += e; }
    float inv = 1.0f / s;
    for (int c = 0; c < 64; ++c) Pb[t * 72 + c] = (_Float16)(Sb[t * 68 + c] * inv);
  }
  __syncthreads();
  // GEMM2: O = P @ M ; B(k,n) = M[k][n] = sMt[n*72 + k]
  {
    int g = (lane >> 4) & 1, n = lane & 15;
    for (int nt = 0; nt < 4; ++nt) {
      v8f acc = {};
#pragma unroll
      for (int ks = 0; ks < 2; ++ks) {
        v16h a = frag_a(Pb, 72, wv * 16, ks * 32, lane);
        v16h b = frag_btr(sMt, 72, ks * 32, nt * 16, lane);
        acc = wmma16(a, b, acc);
      }
#pragma unroll
      for (int v = 0; v < 8; ++v)
        Sb[(wv * 16 + v + 8 * g) * 68 + nt * 16 + n] = acc[v];
    }
  }
  __syncthreads();
  float* Ob = Out + ((size_t)bh * SEQ + (size_t)tile * 128) * DIM;
  for (int i = t; i < 128 * 64; i += 256) Ob[i] = Sb[(i >> 6) * 68 + (i & 63)];
}

// ---------------- host launcher ----------------
extern "C" void kernel_launch(void* const* d_in, const int* in_sizes, int n_in,
                              void* d_out, int out_size, void* d_ws, size_t ws_size,
                              hipStream_t stream) {
  (void)in_sizes; (void)n_in; (void)out_size; (void)ws_size;
  const float* Q = (const float*)d_in[0];
  const float* K = (const float*)d_in[1];
  const float* V = (const float*)d_in[2];
  float* Out = (float*)d_out;

  // workspace carve (~11.6 MB total, L2-resident)
  float* qlm  = (float*)d_ws;                           // BH*4096
  float* klm  = qlm + BH * 4096;                        // BH*4096
  float* A0   = klm + BH * 4096;                        // BH*4096
  float* PINV = A0 + BH * 4096;                         // BH*4096
  float* MM   = PINV + BH * 4096;                       // BH*4096
  float* BVp  = MM + BH * 4096;                         // SPLITS*BH*4096
  float* MLp  = BVp + (size_t)SPLITS * BH * 4096;       // SPLITS*BH*128
  unsigned* coef = (unsigned*)(MLp + (size_t)SPLITS * BH * 128);  // 2 (+pad)
  _Float16* qlmh = (_Float16*)(coef + 4);               // BH*4096 halves
  _Float16* klmh = qlmh + BH * 4096;
  _Float16* MMhT = klmh + BH * 4096;

  k_init<<<1, 32, 0, stream>>>(coef);
  k_landmarks<<<BH, 256, 0, stream>>>(Q, K, qlm, klm, qlmh, klmh);
  k_lm_attn<<<BH, 256, 0, stream>>>(qlm, klm, A0, coef);
  k_pinv<<<BH, 256, 4 * 4096 * sizeof(float), stream>>>(A0, coef, PINV);
  dim3 g4(SPLITS, BH);
  k_bv_partial<<<g4, 256, K4_LDS, stream>>>(K, V, qlmh, BVp, MLp);
  k_bv_combine<<<BH, 256, 0, stream>>>(BVp, MLp, PINV, MM, MMhT);
  dim3 g5(SEQ / 128, BH);
  k_final<<<g5, 256, K5_LDS, stream>>>(Q, klmh, MMhT, Out);
}
